// TriStreamBlock_15848429322500
// MI455X (gfx1250) — compile-verified, matrix operand order checked by reading, not run on hardware
//
#include <hip/hip_runtime.h>
#include <hip/hip_bf16.h>
#include <math.h>

// ---- problem constants (from reference) ----
#define DIMC      192
#define D_STATE   16
#define D_CONV    4
#define D_INNER   384          // EXPAND * DIM
#define DT_RANK   12
#define XPROJ_N   (DT_RANK + 2*D_STATE)   // 44
#define SECRET_D  256
#define TEXT_D    768
#define NB        2
#define LSEQ      4096         // 64*64
#define MROWS     (NB*LSEQ)    // 8192

typedef __attribute__((ext_vector_type(16))) __bf16 v16bf;
typedef __attribute__((ext_vector_type(8)))  __bf16 v8bf;
typedef __attribute__((ext_vector_type(8)))  float  v8f;

__device__ __forceinline__ float silu_f(float x)     { return x / (1.0f + __expf(-x)); }
__device__ __forceinline__ float softplus_f(float x) { return (x > 20.0f) ? x : log1pf(__expf(x)); }

// Fragment loader, bf16 operands already in memory.
// 16-bit A/B 16x32 layout (ISA 7.12.2): lane l holds row (A: m=l%16, B: n=l%16);
//   elems 0..7  -> K = k0 + (l/16)*8 + j
//   elems 8..15 -> K = k0 + 16 + (l/16)*8 + (j-8)
// Each half is 8 bf16 = 16 bytes -> one b128 load; two loads per fragment.
__device__ __forceinline__ v16bf load_frag_bf(const __bf16* __restrict__ row, int k0, int half) {
    const v8bf lo = *reinterpret_cast<const v8bf*>(row + k0 + half * 8);
    const v8bf hi = *reinterpret_cast<const v8bf*>(row + k0 + 16 + half * 8);
    v16bf v;
    #pragma unroll
    for (int i = 0; i < 8; ++i) { v[i] = lo[i]; v[8 + i] = hi[i]; }
    return v;
}

// =====================================================================
// WMMA GEMM: Y[M,N] = A[M,K] @ W[N,K]^T (+ bias) (+ optional silu)
// A, W pre-staged in bf16. One wave32 computes a 32x64 tile
// (2 A frags x 4 B frags -> 8 WMMAs per 32-wide k-chunk).
// K multiple of 32; M multiple of 32. Output f32 (Yf) or bf16 (Yb).
// C/D f32 16x16: lane l holds col n=l%16, rows m = vgpr + 8*(l/16).
// =====================================================================
__global__ void gemm_wmma_kernel(const __bf16* __restrict__ A,
                                 const __bf16* __restrict__ W,
                                 const float* __restrict__ bias,
                                 float*  __restrict__ Yf,
                                 __bf16* __restrict__ Yb,
                                 int M, int N, int K, int epi)
{
    const int wave   = blockIdx.x * (blockDim.x >> 5) + (threadIdx.x >> 5);
    const int tilesN = (N + 63) >> 6;
    const int tilesM = M >> 5;
    if (wave >= tilesM * tilesN) return;   // whole-wave uniform exit

    const int tM   = wave / tilesN;
    const int tN   = wave % tilesN;
    const int lane = threadIdx.x & 31;
    const int half = lane >> 4;
    const int r    = lane & 15;

    const int m0 = tM * 32 + r;
    const __bf16* Arow0 = A + (size_t)m0 * K;
    const __bf16* Arow1 = Arow0 + (size_t)16 * K;

    int  ncol[4];
    bool nval[4];
    const __bf16* Wrow[4];
    #pragma unroll
    for (int j = 0; j < 4; ++j) {
        const int n = tN * 64 + j * 16 + r;
        ncol[j] = n;
        nval[j] = (n < N);
        Wrow[j] = W + (size_t)(nval[j] ? n : 0) * K;
    }

    v8f acc[2][4];
    #pragma unroll
    for (int i = 0; i < 2; ++i)
        #pragma unroll
        for (int j = 0; j < 4; ++j)
            acc[i][j] = (v8f){0.f,0.f,0.f,0.f,0.f,0.f,0.f,0.f};

    for (int k0 = 0; k0 < K; k0 += 32) {
        if (k0 + 32 < K) {
            __builtin_prefetch(Arow0 + k0 + 32, 0, 0);   // global_prefetch_b8
            __builtin_prefetch(Arow1 + k0 + 32, 0, 0);
        }
        const v16bf a0 = load_frag_bf(Arow0, k0, half);
        const v16bf a1 = load_frag_bf(Arow1, k0, half);
        v16bf bf[4];
        #pragma unroll
        for (int j = 0; j < 4; ++j) bf[j] = load_frag_bf(Wrow[j], k0, half);

        #pragma unroll
        for (int j = 0; j < 4; ++j) {
            acc[0][j] = __builtin_amdgcn_wmma_f32_16x16x32_bf16(
                            false, a0, false, bf[j], (short)0, acc[0][j], false, false);
            acc[1][j] = __builtin_amdgcn_wmma_f32_16x16x32_bf16(
                            false, a1, false, bf[j], (short)0, acc[1][j], false, false);
        }
    }

    #pragma unroll
    for (int j = 0; j < 4; ++j) {
        const float bv = (bias != nullptr && nval[j]) ? bias[ncol[j]] : 0.0f;
        #pragma unroll
        for (int i = 0; i < 2; ++i) {
            #pragma unroll
            for (int rr = 0; rr < 8; ++rr) {
                const int mrow = tM * 32 + i * 16 + rr + 8 * half;
                float v = acc[i][j][rr] + bv;
                if (epi == 1) v = silu_f(v);
                if (nval[j]) {
                    const size_t off = (size_t)mrow * N + ncol[j];
                    if (Yf) Yf[off] = v;
                    else    Yb[off] = (__bf16)v;
                }
            }
        }
    }
}

// =====================================================================
// f32 -> bf16 conversion (weight staging)
// =====================================================================
__global__ void cvt_bf16_kernel(const float* __restrict__ in,
                                __bf16* __restrict__ out, int n)
{
    const int i = blockIdx.x * blockDim.x + threadIdx.x;
    if (i < n) out[i] = (__bf16)in[i];
}

// =====================================================================
// c_emb[b,d] = c_txt[b,:] . sem_proj_w[d,:] + sem_proj_b[d]
// =====================================================================
__global__ void cemb_kernel(const float* __restrict__ c_txt,
                            const float* __restrict__ pw,
                            const float* __restrict__ pb,
                            float* __restrict__ out)
{
    const int i = blockIdx.x * blockDim.x + threadIdx.x;
    if (i >= NB * DIMC) return;
    const int b = i / DIMC, d = i % DIMC;
    float s = pb[d];
    const float* ct = c_txt + (size_t)b * TEXT_D;
    const float* wr = pw + (size_t)d * TEXT_D;
    for (int t = 0; t < TEXT_D; ++t) s += ct[t] * wr[t];
    out[i] = s;
}

// =====================================================================
// out_bf[b,l,c] = LN( x_chw[b,c,l]? + add_blc[b,l,c]? + bc[b,c]? ) * w + b
// one block = one (b,l) row, 192 threads; output bf16 (feeds in_proj GEMM)
// =====================================================================
__global__ void add_ln_kernel(const float* __restrict__ x_chw,
                              const float* __restrict__ add_blc,
                              const float* __restrict__ bc,
                              const float* __restrict__ w,
                              const float* __restrict__ b,
                              __bf16* __restrict__ out)
{
    __shared__ float s[DIMC];
    const int bl = blockIdx.x;          // 0 .. MROWS-1
    const int bt = bl / LSEQ;
    const int l  = bl % LSEQ;
    const int c  = threadIdx.x;         // 0 .. 191

    float v = 0.0f;
    if (x_chw)   v += x_chw[((size_t)bt * DIMC + c) * LSEQ + l];
    if (add_blc) v += add_blc[(size_t)bl * DIMC + c];
    if (bc)      v += bc[bt * DIMC + c];

    s[c] = v; __syncthreads();
    for (int off = 96; off >= 3; off >>= 1) {
        if (c < off) s[c] += s[c + off];
        __syncthreads();
    }
    const float mean = (s[0] + s[1] + s[2]) * (1.0f / DIMC);
    __syncthreads();

    const float d = v - mean;
    s[c] = d * d; __syncthreads();
    for (int off = 96; off >= 3; off >>= 1) {
        if (c < off) s[c] += s[c + off];
        __syncthreads();
    }
    const float var = (s[0] + s[1] + s[2]) * (1.0f / DIMC);

    out[(size_t)bl * DIMC + c] = (__bf16)(d * rsqrtf(var + 1e-5f) * w[c] + b[c]);
}

// =====================================================================
// depthwise causal conv (K=4) over L within each batch + bias + silu
// xz: (MROWS, 768), xc = first 384 cols.
// dual output: f32 (scan) + bf16 (x_proj GEMM)
// =====================================================================
__global__ void conv_silu_kernel(const float* __restrict__ xz,
                                 const float* __restrict__ cw,   // (384,1,4)
                                 const float* __restrict__ cb,
                                 float*  __restrict__ out_f,
                                 __bf16* __restrict__ out_b)
{
    const int idx = blockIdx.x * blockDim.x + threadIdx.x;
    if (idx >= MROWS * D_INNER) return;
    const int d  = idx % D_INNER;
    const int bl = idx / D_INNER;
    const int l  = bl % LSEQ;
    const int bt = bl / LSEQ;

    float s = cb[d];
    #pragma unroll
    for (int i = 0; i < D_CONV; ++i) {
        const int t = l - (D_CONV - 1) + i;
        if (t >= 0)
            s += cw[d * D_CONV + i] *
                 xz[((size_t)(bt * LSEQ + t)) * (2 * D_INNER) + d];
    }
    const float v = silu_f(s);
    out_f[idx] = v;
    out_b[idx] = (__bf16)v;
}

// =====================================================================
// dt[b,l,d] = softplus( dbl[b,l,0:12] . dt_w[d,:] + dt_b[d] )
// =====================================================================
__global__ void dt_kernel(const float* __restrict__ dbl,
                          const float* __restrict__ dtw,   // (384,12)
                          const float* __restrict__ dtb,
                          float* __restrict__ out)
{
    const int idx = blockIdx.x * blockDim.x + threadIdx.x;
    if (idx >= MROWS * D_INNER) return;
    const int d  = idx % D_INNER;
    const int bl = idx / D_INNER;
    float s = dtb[d];
    const float* row = dbl + (size_t)bl * XPROJ_N;
    #pragma unroll
    for (int r2 = 0; r2 < DT_RANK; ++r2) s += row[r2] * dtw[d * DT_RANK + r2];
    out[idx] = softplus_f(s);
}

// =====================================================================
// selective-scan: each wave32 handles 2 (b,d) channels, lane%16 = state n.
// h = exp(dt*A)*h + dt*B*x ; y = sum_n h*C + Dp*x ; y *= silu(z)
// cross-state reduce via shfl_xor within 16-lane groups.
// y written as bf16 (only consumer is out_proj GEMM).
// =====================================================================
__global__ void scan_kernel(const float* __restrict__ dt_buf,  // (MROWS,384)
                            const float* __restrict__ dbl,     // (MROWS,44)
                            const float* __restrict__ xc,      // (MROWS,384)
                            const float* __restrict__ xz,      // (MROWS,768) z = cols 384..767
                            const float* __restrict__ A_log,   // (384,16)
                            const float* __restrict__ Dp,      // (384)
                            __bf16* __restrict__ y)            // (MROWS,384) bf16
{
    const int lane = threadIdx.x & 31;
    const int wave = blockIdx.x * (blockDim.x >> 5) + (threadIdx.x >> 5);
    const int ch   = wave * 2 + (lane >> 4);
    if (ch >= NB * D_INNER) return;
    const int bt = ch / D_INNER;
    const int d  = ch % D_INNER;
    const int n  = lane & 15;

    const float Acoef = -__expf(A_log[d * D_STATE + n]);
    const float dcoef = Dp[d];
    float h = 0.0f;

    for (int t = 0; t < LSEQ; ++t) {
        const size_t bl = (size_t)bt * LSEQ + t;
        const float dtv = dt_buf[bl * D_INNER + d];
        const float xv  = xc[bl * D_INNER + d];
        const float Bv  = dbl[bl * XPROJ_N + DT_RANK + n];
        const float Cv  = dbl[bl * XPROJ_N + DT_RANK + D_STATE + n];

        h = __expf(dtv * Acoef) * h + dtv * Bv * xv;

        float part = h * Cv;
        part += __shfl_xor(part, 8, 16);
        part += __shfl_xor(part, 4, 16);
        part += __shfl_xor(part, 2, 16);
        part += __shfl_xor(part, 1, 16);

        if (n == 0) {
            const float zv = xz[bl * (2 * D_INNER) + D_INNER + d];
            y[bl * D_INNER + d] = (__bf16)((part + dcoef * xv) * silu_f(zv));
        }
    }
}

// =====================================================================
// (B,C,L) f32 channel-major -> (B,L,C) bf16 row-major (feeds sp1 GEMM)
// =====================================================================
__global__ void chw2lc_bf_kernel(const float* __restrict__ in,
                                 __bf16* __restrict__ out, int C)
{
    const int idx = blockIdx.x * blockDim.x + threadIdx.x;   // (b,c,l) order
    if (idx >= NB * C * LSEQ) return;
    const int l   = idx % LSEQ;
    const int rem = idx / LSEQ;
    const int c   = rem % C;
    const int bt  = rem / C;
    out[((size_t)(bt * LSEQ + l)) * C + c] = (__bf16)in[idx];
}

// =====================================================================
// pack outputs: out[0] = h_struc as (B,C,H,W)
//               out[1] = (h_tex + f2) as (B,C,H,W)
// =====================================================================
__global__ void finalize_kernel(const float* __restrict__ h_struc,
                                const float* __restrict__ h_tex,
                                const float* __restrict__ f2,
                                float* __restrict__ out)
{
    const int idx = blockIdx.x * blockDim.x + threadIdx.x;   // (b,c,l) order
    if (idx >= NB * DIMC * LSEQ) return;
    const int l   = idx % LSEQ;
    const int rem = idx / LSEQ;
    const int c   = rem % DIMC;
    const int bt  = rem / DIMC;
    const size_t src = ((size_t)(bt * LSEQ + l)) * DIMC + c;
    out[idx] = h_struc[src];
    out[(size_t)NB * DIMC * LSEQ + idx] = h_tex[src] + f2[src];
}

// =====================================================================
// host side
// =====================================================================
struct SsmP {
    const float *in_w, *in_b, *conv_w, *conv_b, *xproj_w,
                *dt_w, *dt_b, *A_log, *Dp, *out_w, *out_b;
};

static void launch_gemm(const __bf16* A, const __bf16* W, const float* bias,
                        float* Yf, __bf16* Yb,
                        int M, int N, int K, int epi, hipStream_t stream)
{
    const int tiles  = (M >> 5) * ((N + 63) >> 6);
    const int blocks = (tiles + 7) / 8;           // 8 waves (256 thr) per block
    gemm_wmma_kernel<<<blocks, 256, 0, stream>>>(A, W, bias, Yf, Yb, M, N, K, epi);
}

static void launch_cvt(const float* in, __bf16* out, int n, hipStream_t stream)
{
    cvt_bf16_kernel<<<(n + 255) / 256, 256, 0, stream>>>(in, out, n);
}

static void run_ssm(const __bf16* ln_bf, const SsmP& p,
                    const __bf16* w_in, const __bf16* w_xp, const __bf16* w_out,
                    float* xz, float* xc_f, __bf16* xc_b, float* dbl,
                    float* dt_buf, __bf16* y_bf, float* out, hipStream_t stream)
{
    const int EW = MROWS * D_INNER;
    // in_proj: (M,192) x (768,192)^T -> (M,768) f32
    launch_gemm(ln_bf, w_in, p.in_b, xz, nullptr, MROWS, 2 * D_INNER, DIMC, 0, stream);
    // depthwise conv + silu on xc half (dual f32/bf16 output)
    conv_silu_kernel<<<(EW + 255) / 256, 256, 0, stream>>>(xz, p.conv_w, p.conv_b, xc_f, xc_b);
    // x_proj: (M,384) x (44,384)^T -> (M,44) f32   [dt | B | C]
    launch_gemm(xc_b, w_xp, nullptr, dbl, nullptr, MROWS, XPROJ_N, D_INNER, 0, stream);
    // dt projection + softplus
    dt_kernel<<<(EW + 255) / 256, 256, 0, stream>>>(dbl, p.dt_w, p.dt_b, dt_buf);
    // sequential selective scan (+ Dp skip + silu(z) gate) -> bf16
    {
        const int waves  = (NB * D_INNER + 1) / 2;  // 384
        const int blocks = (waves + 7) / 8;         // 48
        scan_kernel<<<blocks, 256, 0, stream>>>(dt_buf, dbl, xc_f, xz,
                                                p.A_log, p.Dp, y_bf);
    }
    // out_proj: (M,384) x (192,384)^T -> (M,192) f32
    launch_gemm(y_bf, w_out, p.out_b, out, nullptr, MROWS, DIMC, D_INNER, 0, stream);
}

extern "C" void kernel_launch(void* const* d_in, const int* in_sizes, int n_in,
                              void* d_out, int out_size, void* d_ws, size_t ws_size,
                              hipStream_t stream)
{
    (void)in_sizes; (void)n_in; (void)out_size; (void)ws_size;
    auto in = [&](int i) { return (const float*)d_in[i]; };

    // input layout: 0:x 1:f_sec 2:c_txt, 3..13 sem, 14..24 struc, 25..35 tex,
    // 36/37 sem_norm, 38/39 struc_norm, 40/41 tex_norm,
    // 42/43 sem_proj, 44/45 sp1, 46/47 sp2
    auto ssm = [&](int base) {
        SsmP p;
        p.in_w = in(base + 0);  p.in_b  = in(base + 1);
        p.conv_w = in(base + 2); p.conv_b = in(base + 3);
        p.xproj_w = in(base + 4);
        p.dt_w = in(base + 5);  p.dt_b = in(base + 6);
        p.A_log = in(base + 7); p.Dp = in(base + 8);
        p.out_w = in(base + 9); p.out_b = in(base + 10);
        return p;
    };
    const SsmP sem = ssm(3), struc = ssm(14), tex = ssm(25);

    // ---- workspace carve: f32 region then bf16 region (16B aligned) ----
    float* wsf = (float*)d_ws;
    size_t o = 0;
    auto carveF = [&](size_t nfl) { float* p = wsf + o; o += nfl; return p; };
    float* c_emb   = carveF((size_t)NB * DIMC);            // 384
    float* xz      = carveF((size_t)MROWS * 2 * D_INNER);
    float* xc_f    = carveF((size_t)MROWS * D_INNER);
    float* dbl     = carveF((size_t)MROWS * XPROJ_N);
    float* dt_buf  = carveF((size_t)MROWS * D_INNER);
    float* h_sem   = carveF((size_t)MROWS * DIMC);
    float* h_struc = carveF((size_t)MROWS * DIMC);
    float* h_tex   = carveF((size_t)MROWS * DIMC);
    float* f2      = carveF((size_t)MROWS * DIMC);

    __bf16* wsb = (__bf16*)(wsf + o);
    size_t ob = 0;
    auto carveB = [&](size_t nel) { __bf16* p = wsb + ob; ob += nel; return p; };
    __bf16* ln_bf   = carveB((size_t)MROWS * DIMC);
    __bf16* xc_b    = carveB((size_t)MROWS * D_INNER);
    __bf16* y_bf    = carveB((size_t)MROWS * D_INNER);
    __bf16* fsec_b  = carveB((size_t)MROWS * SECRET_D);
    __bf16* f1_b    = carveB((size_t)MROWS * DIMC);
    // weight arena
    __bf16* w_in[3], *w_xp[3], *w_out[3];
    for (int i = 0; i < 3; ++i) {
        w_in[i]  = carveB((size_t)2 * D_INNER * DIMC);     // 768x192
        w_xp[i]  = carveB((size_t)XPROJ_N * D_INNER);      // 44x384
        w_out[i] = carveB((size_t)DIMC * D_INNER);         // 192x384
    }
    __bf16* w_sp1 = carveB((size_t)DIMC * SECRET_D);       // 192x256
    __bf16* w_sp2 = carveB((size_t)DIMC * DIMC);           // 192x192

    const float* x     = in(0);
    const float* f_sec = in(1);
    const float* c_txt = in(2);

    // ---- stage all GEMM weights to bf16 (once per launch) ----
    const SsmP* ps[3] = { &sem, &struc, &tex };
    for (int i = 0; i < 3; ++i) {
        launch_cvt(ps[i]->in_w,    w_in[i],  2 * D_INNER * DIMC, stream);
        launch_cvt(ps[i]->xproj_w, w_xp[i],  XPROJ_N * D_INNER,  stream);
        launch_cvt(ps[i]->out_w,   w_out[i], DIMC * D_INNER,     stream);
    }
    launch_cvt(in(44), w_sp1, DIMC * SECRET_D, stream);
    launch_cvt(in(46), w_sp2, DIMC * DIMC,     stream);

    // c_emb = c_txt @ sem_proj_w^T + b
    cemb_kernel<<<(NB * DIMC + 255) / 256, 256, 0, stream>>>(c_txt, in(42), in(43), c_emb);

    // ---- stream 1: semantic ----
    add_ln_kernel<<<MROWS, DIMC, 0, stream>>>(x, nullptr, c_emb, in(36), in(37), ln_bf);
    run_ssm(ln_bf, sem, w_in[0], w_xp[0], w_out[0],
            xz, xc_f, xc_b, dbl, dt_buf, y_bf, h_sem, stream);

    // ---- stream 2: structural ----
    add_ln_kernel<<<MROWS, DIMC, 0, stream>>>(x, h_sem, nullptr, in(38), in(39), ln_bf);
    run_ssm(ln_bf, struc, w_in[1], w_xp[1], w_out[1],
            xz, xc_f, xc_b, dbl, dt_buf, y_bf, h_struc, stream);

    // ---- stream 3: texture ----
    add_ln_kernel<<<MROWS, DIMC, 0, stream>>>(nullptr, h_struc, nullptr, in(40), in(41), ln_bf);
    run_ssm(ln_bf, tex, w_in[2], w_xp[2], w_out[2],
            xz, xc_f, xc_b, dbl, dt_buf, y_bf, h_tex, stream);

    // ---- secret modulation path ----
    {
        const int n = NB * SECRET_D * LSEQ;
        chw2lc_bf_kernel<<<(n + 255) / 256, 256, 0, stream>>>(f_sec, fsec_b, SECRET_D);
    }
    // sp1 (+silu) -> bf16 directly (only consumer is sp2 GEMM)
    launch_gemm(fsec_b, w_sp1, in(45), nullptr, f1_b, MROWS, DIMC, SECRET_D, 1, stream);
    launch_gemm(f1_b,   w_sp2, in(47), f2, nullptr, MROWS, DIMC, DIMC,     0, stream);

    // ---- pack outputs ----
    {
        const int n = NB * DIMC * LSEQ;
        finalize_kernel<<<(n + 255) / 256, 256, 0, stream>>>(h_struc, h_tex, f2, (float*)d_out);
    }
}
